// Infer_CascadeREDNet_51599737094659
// MI455X (gfx1250) — matrix operand order repaired and checked by reading, not compiled, not run on hardware
//
#include <hip/hip_runtime.h>
#include <hip/hip_bf16.h>
#include <math.h>

typedef __attribute__((ext_vector_type(16))) __bf16 v16bf;
typedef __attribute__((ext_vector_type(8)))  float  v8f;

#define ACT_LIN  0
#define ACT_RELU 1
#define ACT_SIG  2
#define ACT_TANH 3

// ---------------------------------------------------------------------------
// Pack f32 OIHW (or transposed IOHW, flipped) 3x3 weights into the CDNA5
// WMMA bf16 A-fragment layout, using this file's contiguous K order (16x32 tile per wave):
//   lane L: M = mtile*16 + (L&15), khalf = L>>4
//   j in [0,16): Klocal = khalf*16 + j
// ---------------------------------------------------------------------------
__global__ void pack_w_kernel(const float* __restrict__ w, __bf16* __restrict__ out,
                              int Cout, int Cin, int Ktot, int ktiles, int tf, int total)
{
    int e = blockIdx.x * blockDim.x + threadIdx.x;
    if (e >= total) return;
    int j     = e & 15;
    int lane  = (e >> 4) & 31;
    int tile  = e >> 9;
    int kt    = tile % ktiles;
    int mtile = tile / ktiles;
    int m     = mtile * 16 + (lane & 15);
    int khalf = lane >> 4;
    int klocal = khalf * 16 + j;
    int k = kt * 32 + klocal;
    float val = 0.f;
    if (m < Cout && k < Ktot) {
        int ci = k / 9, r = k - ci * 9, ky = r / 3, kx = r - ky * 3;
        if (!tf) val = w[((m * Cin + ci) * 3 + ky) * 3 + kx];              // (Cout,Cin,3,3)
        else     val = w[((ci * Cout + m) * 3 + (2 - ky)) * 3 + (2 - kx)]; // (Cin,Cout,3,3) flip
    }
    out[e] = (__bf16)val;
}

// ---------------------------------------------------------------------------
// 3x3 conv / 2x-transposed-conv as implicit GEMM on v_wmma_f32_16x16x32_bf16.
//   M = Cout (MT 16-row tiles, all done by one wave reusing B), N = 16 output
//   pixels, K = (C1+C2)*9 padded to 32.
// B fragment per lane: column = lane&15, 16 contiguous K starting at
// kt*32 + (lane>>4)*16.  (ci,ky,kx) tracked incrementally -> no divisions.
// MODE 0: stride-s conv pad 1.  MODE 1: 2x transposed conv (pad 1, outpad 1).
// X = concat(X1[C1], X2[C2]); optional elementwise multiplier on X2 (GRU r*h).
// ---------------------------------------------------------------------------
template<int MT, int MODE>
__global__ __launch_bounds__(128)
void conv_wmma_kernel(const float* __restrict__ X1, int C1,
                      const float* __restrict__ X2, int C2,
                      const float* __restrict__ Mul2,
                      const __bf16* __restrict__ Wp,
                      const float* __restrict__ bias,
                      float* __restrict__ Y,
                      int Cout, int Hin, int Win, int Hout, int Wout,
                      int stride, int act, int ktiles, int ktFull)
{
    const int lane  = threadIdx.x;                        // 0..31
    const int ntile = blockIdx.x * blockDim.y + threadIdx.y;
    const int npix  = Hout * Wout;
    const int col   = ntile * 16 + (lane & 15);
    const bool colV = (col < npix);
    const int oy    = colV ? (col / Wout) : 0;
    const int ox    = colV ? (col - (col / Wout) * Wout) : 0;
    const int khalf = lane >> 4;
    const int Ctot  = C1 + C2;
    const size_t planeIn = (size_t)Hin * Win;
    const v16bf* Wv = (const v16bf*)Wp;
    const bool hasMul = (Mul2 != nullptr);

    // pre-biased input anchors for this output pixel
    const int oyb = (MODE == 0) ? (oy * stride - 1) : (oy - 1);
    const int oxb = (MODE == 0) ? (ox * stride - 1) : (ox - 1);

    v8f acc[MT];
#pragma unroll
    for (int mt = 0; mt < MT; ++mt)
        acc[mt] = v8f{0.f, 0.f, 0.f, 0.f, 0.f, 0.f, 0.f, 0.f};

    // incremental (ci,ky,kx) for this lane's K stream: k = kt*32 + khalf*16 + j
    // khalf==0 -> k0=0  -> (0,0,0);  khalf==1 -> k0=16=9+7 -> (1,2,1)
    int ci = khalf;
    int ky = khalf * 2;
    int kx = khalf;

    for (int kt = 0; kt < ktiles; ++kt) {
        const bool tail = (kt >= ktFull);
        v16bf b;
#pragma unroll
        for (int j = 0; j < 16; ++j) {
            float v = 0.f;
            bool okc = colV && (!tail || ci < Ctot);
            if (okc) {
                int iy, ix; bool ok;
                if (MODE == 0) {
                    iy = oyb + ky; ix = oxb + kx;
                    ok = ((unsigned)iy < (unsigned)Hin) && ((unsigned)ix < (unsigned)Win);
                } else {
                    int ty = oyb + ky, tx = oxb + kx;
                    ok = ((ty | tx) >= 0) && ((ty & 1) == 0) && ((tx & 1) == 0);
                    iy = ty >> 1; ix = tx >> 1;
                    ok = ok && (iy < Hin) && (ix < Win);
                }
                if (ok) {
                    if (ci < C1) {
                        v = X1[(size_t)ci * planeIn + iy * Win + ix];
                    } else {
                        size_t o2 = (size_t)(ci - C1) * planeIn + iy * Win + ix;
                        v = X2[o2];
                        if (hasMul) v *= Mul2[o2];
                    }
                }
            }
            b[j] = (__bf16)v;
            // advance one K step
            if (++kx == 3) { kx = 0; if (++ky == 3) { ky = 0; ++ci; } }
        }
        // skip the other khalf's 16 K values to land on next tile's start
        {
            int t = ky * 3 + kx + 16;            // in [16,24]
            int add = (t >= 18) ? 2 : 1;
            ci += add; t -= 9 * add;             // t in [0,8]
            ky = (t >= 6) ? 2 : ((t >= 3) ? 1 : 0);
            kx = t - 3 * ky;
        }
        if (kt + 1 < ktiles)
            __builtin_prefetch(&Wv[(size_t)(kt + 1) * 32 + lane], 0, 3);
        // MT independent WMMAs reuse the same B fragment (no D->A/B hazards)
#pragma unroll
        for (int mt = 0; mt < MT; ++mt) {
            v16bf a = Wv[(size_t)(mt * ktiles + kt) * 32 + lane];
            acc[mt] = __builtin_amdgcn_wmma_f32_16x16x32_bf16(false, a, false, b,
                                                              (short)0, acc[mt],
                                                              false, false);
        }
    }

#pragma unroll
    for (int mt = 0; mt < MT; ++mt) {
#pragma unroll
        for (int vr = 0; vr < 8; ++vr) {
            int m = mt * 16 + vr + 8 * khalf;   // C/D layout: M = vgpr + 8*khalf
            if (m < Cout && colV) {
                float y = acc[mt][vr] + bias[m];
                if      (act == ACT_RELU) y = fmaxf(y, 0.f);
                else if (act == ACT_SIG)  y = 1.f / (1.f + __expf(-y));
                else if (act == ACT_TANH) y = tanhf(y);
                Y[(size_t)m * npix + col] = y;
            }
        }
    }
}

// ---------------------------------------------------------------------------
// 1-thread: inv(ref_proj) (Gauss-Jordan), then rot/trans of src_proj*inv.
// ---------------------------------------------------------------------------
__global__ void prep_proj_kernel(const float* __restrict__ projs, float* __restrict__ RT, int V)
{
    if (threadIdx.x != 0 || blockIdx.x != 0) return;
    float A[4][8];
    for (int r = 0; r < 4; ++r)
        for (int c = 0; c < 4; ++c) { A[r][c] = projs[r * 4 + c]; A[r][c + 4] = (r == c) ? 1.f : 0.f; }
    for (int c = 0; c < 4; ++c) {
        int piv = c; float best = fabsf(A[c][c]);
        for (int r = c + 1; r < 4; ++r) { float t = fabsf(A[r][c]); if (t > best) { best = t; piv = r; } }
        if (piv != c) for (int k = 0; k < 8; ++k) { float t = A[c][k]; A[c][k] = A[piv][k]; A[piv][k] = t; }
        float inv = 1.f / A[c][c];
        for (int k = 0; k < 8; ++k) A[c][k] *= inv;
        for (int r = 0; r < 4; ++r) {
            if (r == c) continue;
            float f = A[r][c];
            for (int k = 0; k < 8; ++k) A[r][k] -= f * A[c][k];
        }
    }
    for (int v = 1; v < V; ++v) {
        const float* P = projs + v * 16;
        float* o = RT + (v - 1) * 12;
        for (int r = 0; r < 3; ++r) {
            float M[4];
            for (int c = 0; c < 4; ++c) {
                float s = 0.f;
                for (int k = 0; k < 4; ++k) s += P[r * 4 + k] * A[k][c + 4];
                M[c] = s;
            }
            o[r * 3 + 0] = M[0]; o[r * 3 + 1] = M[1]; o[r * 3 + 2] = M[2];
            o[9 + r] = M[3];
        }
    }
}

// ---------------------------------------------------------------------------
// Homography warp + variance cost volume (writes nc = -variance directly).
// ---------------------------------------------------------------------------
__global__ void warp_var_kernel(const float* __restrict__ feat, const float* __restrict__ RT,
                                const float* __restrict__ depthv, int d,
                                float* __restrict__ nc, int C, int H, int W, int V)
{
    int p = blockIdx.x * blockDim.x + threadIdx.x;
    int HW = H * W;
    if (p >= HW) return;
    float xx = (float)(p % W);
    float yy = (float)(p / W);
    float depth = depthv[d];

    float wgt[3][4];
    int   idx[3][4];
    int nv = V - 1; if (nv > 3) nv = 3;
    for (int v = 0; v < nv; ++v) {
        const float* r = RT + v * 12;
        float X  = (r[0] * xx + r[1] * yy + r[2]) * depth + r[9];
        float Yc = (r[3] * xx + r[4] * yy + r[5]) * depth + r[10];
        float Z  = (r[6] * xx + r[7] * yy + r[8]) * depth + r[11];
        float px = X / Z, py = Yc / Z;
        float x0 = floorf(px), y0 = floorf(py);
        float wx1 = px - x0, wx0 = 1.f - wx1;
        float wy1 = py - y0, wy0 = 1.f - wy1;
        float wtab[4] = { wx0 * wy0, wx1 * wy0, wx0 * wy1, wx1 * wy1 };
        for (int t = 0; t < 4; ++t) {
            float xi = x0 + (float)(t & 1);
            float yi = y0 + (float)(t >> 1);
            float valid = (xi >= 0.f && xi <= (float)(W - 1) &&
                           yi >= 0.f && yi <= (float)(H - 1)) ? 1.f : 0.f;
            float xc = fminf(fmaxf(xi, 0.f), (float)(W - 1));
            float yc = fminf(fmaxf(yi, 0.f), (float)(H - 1));
            idx[v][t] = (int)yc * W + (int)xc;
            wgt[v][t] = wtab[t] * valid;
        }
    }
    float invV = 1.f / (float)V;
    for (int c = 0; c < C; ++c) {
        float f  = feat[(size_t)c * HW + p];   // ref view
        float vs = f, vsq = f * f;
        for (int v = 0; v < nv; ++v) {
            const float* fc = feat + ((size_t)(v + 1) * C + c) * HW;
            float s = wgt[v][0] * fc[idx[v][0]] + wgt[v][1] * fc[idx[v][1]]
                    + wgt[v][2] * fc[idx[v][2]] + wgt[v][3] * fc[idx[v][3]];
            vs += s; vsq += s * s;
        }
        float mean = vs * invV;
        nc[(size_t)c * HW + p] = -(vsq * invV - mean * mean);
    }
}

// ---------------------------------------------------------------------------
// Elementwise helpers
// ---------------------------------------------------------------------------
__global__ void fill0_kernel(float* p, int n)
{ int i = blockIdx.x * blockDim.x + threadIdx.x; if (i < n) p[i] = 0.f; }

__global__ void add_kernel(const float* a, const float* b, float* o, int n)
{ int i = blockIdx.x * blockDim.x + threadIdx.x; if (i < n) o[i] = a[i] + b[i]; }

__global__ void gru_mix_kernel(float* h, const float* u, const float* c, int n)
{ int i = blockIdx.x * blockDim.x + threadIdx.x;
  if (i < n) { float uu = u[i]; h[i] = uu * h[i] + (1.f - uu) * c[i]; } }

__global__ void accum_kernel(const float* reg, const float* depthv, int d,
                             float* esum, float* dimg, float* mpr, int n)
{
    int i = blockIdx.x * blockDim.x + threadIdx.x;
    if (i < n) {
        float pr = __expf(reg[i]);
        esum[i] += pr;
        dimg[i] += depthv[d] * pr;
        mpr[i]   = fmaxf(mpr[i], pr);
    }
}

__global__ void final_kernel(const float* dimg, const float* esum, const float* mpr,
                             float* out, int n)
{
    int i = blockIdx.x * blockDim.x + threadIdx.x;
    if (i < n) {
        float es = esum[i] + 1e-10f;
        out[i]     = dimg[i] / es;
        out[n + i] = mpr[i]  / es;
    }
}

// ---------------------------------------------------------------------------
// Host orchestration
// ---------------------------------------------------------------------------
extern "C" void kernel_launch(void* const* d_in, const int* in_sizes, int n_in,
                              void* d_out, int out_size, void* d_ws, size_t ws_size,
                              hipStream_t stream)
{
    (void)n_in; (void)out_size; (void)ws_size;
    const int Bc = 8;                       // base channels
    const int C = 32, H = 128, W = 160, V = 3;
    const int HW = H * W;
    const int Dn = in_sizes[2];             // number of depth planes

    const float* features = (const float*)d_in[0];
    const float* projm    = (const float*)d_in[1];
    const float* depthv   = (const float*)d_in[2];
    // d_in[3] = num_depth (scalar, unused)

    // ---- bump allocator over d_ws (deterministic every call) ----
    char* base = (char*)d_ws;
    size_t off = 0;
    auto alloc = [&](size_t bytes) -> void* {
        void* p = (void*)(base + off);
        off = (off + bytes + 255) & ~(size_t)255;
        return p;
    };

    // ---- pack all weights into WMMA bf16 A-fragment layout ----
    auto pack = [&](int widx, int Cout, int Cin, int tf) -> __bf16* {
        int Ktot = Cin * 9;
        int ktiles = (Ktot + 31) / 32;
        int mtiles = (Cout + 15) / 16;
        int total  = mtiles * ktiles * 512;
        __bf16* dst = (__bf16*)alloc((size_t)total * sizeof(__bf16));
        pack_w_kernel<<<(total + 255) / 256, 256, 0, stream>>>(
            (const float*)d_in[widx], dst, Cout, Cin, Ktot, ktiles, tf, total);
        return dst;
    };

    __bf16* pk_c1  = pack(4,  2 * Bc, C,       0);
    __bf16* pk_c2  = pack(6,  4 * Bc, 2 * Bc,  0);
    __bf16* pk_c3  = pack(8,  8 * Bc, 4 * Bc,  0);
    __bf16* pk_g1r = pack(10, Bc,     C + Bc,  0);
    __bf16* pk_g1u = pack(12, Bc,     C + Bc,  0);
    __bf16* pk_g1c = pack(14, Bc,     C + Bc,  0);
    __bf16* pk_g2r = pack(16, 2 * Bc, 4 * Bc,  0);
    __bf16* pk_g2u = pack(18, 2 * Bc, 4 * Bc,  0);
    __bf16* pk_g2c = pack(20, 2 * Bc, 4 * Bc,  0);
    __bf16* pk_g3r = pack(22, 4 * Bc, 8 * Bc,  0);
    __bf16* pk_g3u = pack(24, 4 * Bc, 8 * Bc,  0);
    __bf16* pk_g3c = pack(26, 4 * Bc, 8 * Bc,  0);
    __bf16* pk_g4r = pack(28, 8 * Bc, 16 * Bc, 0);
    __bf16* pk_g4u = pack(30, 8 * Bc, 16 * Bc, 0);
    __bf16* pk_g4c = pack(32, 8 * Bc, 16 * Bc, 0);
    __bf16* pk_u3  = pack(34, 4 * Bc, 8 * Bc,  1);   // transposed+flipped
    __bf16* pk_u2  = pack(36, 2 * Bc, 4 * Bc,  1);
    __bf16* pk_u1  = pack(38, Bc,     2 * Bc,  1);
    __bf16* pk_ow  = pack(40, 1,      Bc,      1);

    const float* bc1 = (const float*)d_in[5];
    const float* bc2 = (const float*)d_in[7];
    const float* bc3 = (const float*)d_in[9];
    const float* bu3 = (const float*)d_in[35];
    const float* bu2 = (const float*)d_in[37];
    const float* bu1 = (const float*)d_in[39];
    const float* bow = (const float*)d_in[41];

    // ---- activation / state buffers (f32, channel-major) ----
    float* RT   = (float*)alloc(24 * sizeof(float));
    float* nc   = (float*)alloc((size_t)C * HW * sizeof(float));
    float* a1   = (float*)alloc((size_t)2 * Bc * (HW / 4)  * sizeof(float));
    float* a2   = (float*)alloc((size_t)4 * Bc * (HW / 16) * sizeof(float));
    float* a3   = (float*)alloc((size_t)8 * Bc * (HW / 64) * sizeof(float));
    float* s1   = (float*)alloc((size_t)Bc * HW * sizeof(float));
    float* s2   = (float*)alloc((size_t)2 * Bc * (HW / 4)  * sizeof(float));
    float* s3   = (float*)alloc((size_t)4 * Bc * (HW / 16) * sizeof(float));
    float* s4   = (float*)alloc((size_t)8 * Bc * (HW / 64) * sizeof(float));
    float* rbuf = (float*)alloc((size_t)Bc * HW * sizeof(float));
    float* ubuf = (float*)alloc((size_t)Bc * HW * sizeof(float));
    float* cbuf = (float*)alloc((size_t)Bc * HW * sizeof(float));
    float* u3   = (float*)alloc((size_t)4 * Bc * (HW / 16) * sizeof(float));
    float* u2   = (float*)alloc((size_t)2 * Bc * (HW / 4)  * sizeof(float));
    float* u1   = (float*)alloc((size_t)Bc * HW * sizeof(float));
    float* sum  = (float*)alloc((size_t)Bc * HW * sizeof(float));
    float* reg  = (float*)alloc((size_t)HW * sizeof(float));
    float* esum = (float*)alloc((size_t)HW * sizeof(float));
    float* dimg = (float*)alloc((size_t)HW * sizeof(float));
    float* mpr  = (float*)alloc((size_t)HW * sizeof(float));

    auto zero = [&](float* p, int n) {
        fill0_kernel<<<(n + 255) / 256, 256, 0, stream>>>(p, n);
    };
    zero(s1, Bc * HW);            zero(s2, 2 * Bc * (HW / 4));
    zero(s3, 4 * Bc * (HW / 16)); zero(s4, 8 * Bc * (HW / 64));
    zero(esum, HW); zero(dimg, HW); zero(mpr, HW);

    prep_proj_kernel<<<1, 1, 0, stream>>>(projm, RT, V);

    // ---- conv launcher: one wave handles all M-tiles (B-fragment reuse) ----
    auto conv = [&](const float* X1, int C1, const float* X2, int C2, const float* mul,
                    const __bf16* Wp, const float* bias, float* Yo, int Cout,
                    int Hin, int Win, int Hout, int Wout, int stride, int mode, int act) {
        int Ktot   = (C1 + C2) * 9;
        int ktiles = (Ktot + 31) / 32;
        int ktFull = Ktot / 32;
        int mtiles = (Cout + 15) / 16;
        int ntiles = (Hout * Wout + 15) / 16;
        dim3 grid((ntiles + 3) / 4), block(32, 4);
#define LAUNCH_CONV(MT, MD)                                                         \
        conv_wmma_kernel<MT, MD><<<grid, block, 0, stream>>>(                       \
            X1, C1, X2, C2, mul, Wp, bias, Yo, Cout,                                \
            Hin, Win, Hout, Wout, stride, act, ktiles, ktFull)
        if (mode == 0) {
            if      (mtiles == 1) LAUNCH_CONV(1, 0);
            else if (mtiles == 2) LAUNCH_CONV(2, 0);
            else                  LAUNCH_CONV(4, 0);
        } else {
            if      (mtiles == 1) LAUNCH_CONV(1, 1);
            else if (mtiles == 2) LAUNCH_CONV(2, 1);
            else                  LAUNCH_CONV(4, 1);
        }
#undef LAUNCH_CONV
    };

    // ---- GRU: r=sig(conv[x,h]), u=sig(conv[x,h]), c=tanh(conv[x,r*h]), h=u*h+(1-u)*c
    auto gru = [&](const float* x, int cx, float* h, int ch,
                   const __bf16* pr, const __bf16* pu, const __bf16* pc, int gi,
                   int Hh, int Ww) {
        const float* br  = (const float*)d_in[gi + 1];
        const float* bu  = (const float*)d_in[gi + 3];
        const float* bcv = (const float*)d_in[gi + 5];
        conv(x, cx, h, ch, nullptr, pr, br,  rbuf, ch, Hh, Ww, Hh, Ww, 1, 0, ACT_SIG);
        conv(x, cx, h, ch, nullptr, pu, bu,  ubuf, ch, Hh, Ww, Hh, Ww, 1, 0, ACT_SIG);
        conv(x, cx, h, ch, rbuf,    pc, bcv, cbuf, ch, Hh, Ww, Hh, Ww, 1, 0, ACT_TANH);
        int n = ch * Hh * Ww;
        gru_mix_kernel<<<(n + 255) / 256, 256, 0, stream>>>(h, ubuf, cbuf, n);
    };
    auto addb = [&](const float* a, const float* b, float* o, int n) {
        add_kernel<<<(n + 255) / 256, 256, 0, stream>>>(a, b, o, n);
    };

    // ---- sequential depth scan (GRU recurrence across planes) ----
    for (int d = 0; d < Dn; ++d) {
        warp_var_kernel<<<(HW + 255) / 256, 256, 0, stream>>>(
            features, RT, depthv, d, nc, C, H, W, V);

        conv(nc, C,      nullptr, 0, nullptr, pk_c1, bc1, a1, 2 * Bc, H,     W,     H / 2, W / 2, 2, 0, ACT_RELU);
        conv(a1, 2 * Bc, nullptr, 0, nullptr, pk_c2, bc2, a2, 4 * Bc, H / 2, W / 2, H / 4, W / 4, 2, 0, ACT_RELU);
        conv(a2, 4 * Bc, nullptr, 0, nullptr, pk_c3, bc3, a3, 8 * Bc, H / 4, W / 4, H / 8, W / 8, 2, 0, ACT_RELU);

        gru(a3, 8 * Bc, s4, 8 * Bc, pk_g4r, pk_g4u, pk_g4c, 28, H / 8, W / 8);
        conv(s4, 8 * Bc, nullptr, 0, nullptr, pk_u3, bu3, u3, 4 * Bc, H / 8, W / 8, H / 4, W / 4, 1, 1, ACT_RELU);

        gru(a2, 4 * Bc, s3, 4 * Bc, pk_g3r, pk_g3u, pk_g3c, 22, H / 4, W / 4);
        addb(u3, s3, sum, 4 * Bc * (HW / 16));
        conv(sum, 4 * Bc, nullptr, 0, nullptr, pk_u2, bu2, u2, 2 * Bc, H / 4, W / 4, H / 2, W / 2, 1, 1, ACT_RELU);

        gru(a1, 2 * Bc, s2, 2 * Bc, pk_g2r, pk_g2u, pk_g2c, 16, H / 2, W / 2);
        addb(u2, s2, sum, 2 * Bc * (HW / 4));
        conv(sum, 2 * Bc, nullptr, 0, nullptr, pk_u1, bu1, u1, Bc, H / 2, W / 2, H, W, 1, 1, ACT_RELU);

        gru(nc, C, s1, Bc, pk_g1r, pk_g1u, pk_g1c, 10, H, W);
        addb(u1, s1, sum, Bc * HW);
        conv(sum, Bc, nullptr, 0, nullptr, pk_ow, bow, reg, 1, H, W, H, W, 1, 0, ACT_LIN);

        accum_kernel<<<(HW + 255) / 256, 256, 0, stream>>>(reg, depthv, d, esum, dimg, mpr, HW);
    }

    final_kernel<<<(HW + 255) / 256, 256, 0, stream>>>(dimg, esum, mpr, (float*)d_out, HW);
}
